// InformedRnnSenderFixedLengthGS_48704929136907
// MI455X (gfx1250) — matrix operand order repaired and checked by reading, not compiled, not run on hardware
//
#include <hip/hip_runtime.h>

#define B_   8
#define D_   2048
#define E_   512
#define H_   512
#define V_   32000
#define T_   32
#define G3   1536
#define KSLICES 50
#define KC_PER_SLICE 20   // 20*32 = 640 K per slice; 50*640 = 32000

typedef __attribute__((ext_vector_type(16))) __bf16 v16bf;
typedef __attribute__((ext_vector_type(8)))  __bf16 v8bf;
typedef __attribute__((ext_vector_type(8)))  float  v8f;
typedef __attribute__((ext_vector_type(2)))  float  v2f;

// ---- WMMA wrappers -------------------------------------------------------
__device__ __forceinline__ v8f wmma_bf16(v16bf a, v16bf b, v8f c) {
  return __builtin_amdgcn_wmma_f32_16x16x32_bf16(false, a, false, b, (short)0, c, false, false);
}
__device__ __forceinline__ v8f wmma_f32(v2f a, v2f b, v8f c) {
  return __builtin_amdgcn_wmma_f32_16x16x4_f32(false, a, false, b, (short)0, c, false, false);
}

// ---- bf16 fragment loads (layouts per CDNA5 ISA §7.12.2) ------
// A (16x32, row-major src[ld]): lane m=L&15, kg=L>>4.
// VGPR0-3 = K[k0+8kg .. +7], VGPR4-7 = K[k0+16+8kg .. +7]  (each 16B contiguous)
__device__ __forceinline__ v16bf load_a16(const __bf16* src, int ld, int m, int kg, int k0) {
  const __bf16* p0 = src + (size_t)m * ld + k0 + 8 * kg;
  v8bf lo = *(const v8bf*)p0;
  v8bf hi = *(const v8bf*)(p0 + 16);
  v16bf a;
#pragma unroll
  for (int i = 0; i < 8; ++i) { a[i] = lo[i]; a[i + 8] = hi[i]; }
  return a;
}
// B (32x16) from row-major W[N][K] (computing X @ W^T): lane col n,
// VGPR0-3 = K[k0+8kg .. +7], VGPR4-7 = K[k0+16+8kg .. +7] (matches A).
__device__ __forceinline__ v16bf load_b16(const __bf16* w, int ld, int n, int kg, int k0) {
  const __bf16* p = w + (size_t)n * ld + k0 + 8 * kg;
  v8bf lo = *(const v8bf*)p;
  v8bf hi = *(const v8bf*)(p + 16);
  v16bf b;
#pragma unroll
  for (int i = 0; i < 8; ++i) { b[i] = lo[i]; b[i + 8] = hi[i]; }
  return b;
}

// ---- prep kernels --------------------------------------------------------
__global__ void cvt_bf16(const float* __restrict__ src, __bf16* __restrict__ dst, int n) {
  int i = blockIdx.x * blockDim.x + threadIdx.x;
  if (i < n) dst[i] = (__bf16)src[i];
}
// emb_in [V,E] -> transposed bf16 [E][V] so the e_next B-fragment is contiguous in K(=v)
__global__ void tcvt_embin(const float* __restrict__ src, __bf16* __restrict__ dst) {
  int i = blockIdx.x * blockDim.x + threadIdx.x;   // over E_*V_
  if (i < E_ * V_) {
    int e = i / V_, v = i % V_;
    dst[i] = (__bf16)src[(size_t)v * E_ + e];
  }
}
__global__ void pad_x(const float* __restrict__ x, float* __restrict__ h0) {
  int i = blockIdx.x * blockDim.x + threadIdx.x;   // 16*2048
  int r = i >> 11, c = i & 2047;
  h0[i] = (r < B_) ? x[r * D_ + c] : 0.f;
}
__global__ void init_state(const float* __restrict__ ph0, float* __restrict__ ht,
                           __bf16* __restrict__ htb, __bf16* __restrict__ etb) {
  int i = blockIdx.x * blockDim.x + threadIdx.x;   // 16*512
  int r = i >> 9, c = i & 511;
  float v = (r < B_) ? ph0[c] : 0.f;
  ht[i] = v; htb[i] = (__bf16)v; etb[i] = (__bf16)0.f;
}

// ---- encoder: f32 WMMA (16x16x4), dst = act(src @ W^T + bias) ------------
__global__ void dense_f32_wmma(const float* __restrict__ src, const float* __restrict__ W,
                               const float* __restrict__ bias, float* __restrict__ dst,
                               int N, int K, int apply_tanh) {
  int lane = threadIdx.x & 31, wave = threadIdx.x >> 5;
  int n0 = (blockIdx.x * 8 + wave) * 16;
  int mm = lane & 15, hg = lane >> 4;
  int n = n0 + mm;
  v8f c = {};
  for (int k0 = 0; k0 < K; k0 += 4) {
    int kp = k0 + 2 * hg;                  // A: lane half-group selects K pair
    v2f a = *(const v2f*)(src + (size_t)mm * K + kp);
    v2f b = *(const v2f*)(W + (size_t)n * K + kp);
    c = wmma_f32(a, b, c);
  }
  float bb = bias[n];
#pragma unroll
  for (int vg = 0; vg < 8; ++vg) {
    int m = vg + 8 * hg;                   // D: lane=col, VGPR=row%8
    float val = c[vg] + bb;
    if (apply_tanh) val = tanhf(val);
    dst[(size_t)m * N + n] = val;
  }
}

// ---- GRU gate GEMMs: Gi = E@Wih^T + b_ih ; Gh = H@Whh^T + b_hh -----------
__global__ void gru_matmul(const __bf16* __restrict__ etb, const __bf16* __restrict__ htb,
                           const __bf16* __restrict__ wih, const __bf16* __restrict__ whh,
                           const float* __restrict__ bih, const float* __restrict__ bhh,
                           float* __restrict__ gi, float* __restrict__ gh) {
  int lane = threadIdx.x & 31, wave = threadIdx.x >> 5;
  int n0 = (blockIdx.x * 8 + wave) * 16;   // 96 tiles over 3H=1536
  int mm = lane & 15, hg = lane >> 4;
  int n = n0 + mm;
  v8f ci = {}, ch = {};
  for (int k0 = 0; k0 < E_; k0 += 32) {
    v16bf ae = load_a16(etb, E_, mm, hg, k0);
    v16bf ah = load_a16(htb, H_, mm, hg, k0);
    v16bf bi = load_b16(wih, E_, n, hg, k0);
    v16bf bh = load_b16(whh, H_, n, hg, k0);
    ci = wmma_bf16(ae, bi, ci);
    ch = wmma_bf16(ah, bh, ch);
  }
  float bbi = bih[n], bbh = bhh[n];
#pragma unroll
  for (int vg = 0; vg < 8; ++vg) {
    int m = vg + 8 * hg;
    if (m < B_) { gi[m * G3 + n] = ci[vg] + bbi; gh[m * G3 + n] = ch[vg] + bbh; }
  }
}

__global__ void gru_pointwise(const float* __restrict__ gi, const float* __restrict__ gh,
                              float* __restrict__ ht, __bf16* __restrict__ htb) {
  int i = blockIdx.x * blockDim.x + threadIdx.x;   // 8*512
  int b = i >> 9, j = i & 511;
  int base = b * G3 + j;
  float r = 1.f / (1.f + __expf(-(gi[base] + gh[base])));
  float z = 1.f / (1.f + __expf(-(gi[base + H_] + gh[base + H_])));
  float nn = tanhf(gi[base + 2 * H_] + r * gh[base + 2 * H_]);
  float hp = ht[i];
  float h = (1.f - z) * nn + z * hp;
  ht[i] = h; htb[i] = (__bf16)h;
}

// ---- logits = h_t @ emb_out^T + gumbel[t]  (dominant stream, bf16 WMMA) --
__global__ void logits_wmma(const __bf16* __restrict__ htb, const __bf16* __restrict__ embout,
                            const float* __restrict__ g_t, float* __restrict__ logits) {
  int lane = threadIdx.x & 31, wave = threadIdx.x >> 5;
  int n0 = (blockIdx.x * 8 + wave) * 16;   // 2000 tiles over V
  int mm = lane & 15, hg = lane >> 4;
  int n = n0 + mm;
  v8f c = {};
#pragma unroll 4
  for (int k0 = 0; k0 < H_; k0 += 32) {
    v16bf a = load_a16(htb, H_, mm, hg, k0);
    v16bf b = load_b16(embout, H_, n, hg, k0);
    c = wmma_bf16(a, b, c);
  }
#pragma unroll
  for (int vg = 0; vg < 8; ++vg) {
    int m = vg + 8 * hg;
    if (m < B_) logits[(size_t)m * V_ + n] = c[vg] + g_t[(size_t)m * V_ + n];
  }
}

// ---- online softmax row reduction (one block per batch row) --------------
__global__ void softmax_reduce(const float* __restrict__ logits, float* __restrict__ red) {
  __shared__ float sx[256], sm[256];
  int b = blockIdx.x, tid = threadIdx.x;
  float mx = -3.4e38f, s = 0.f;
  for (int v = tid; v < V_; v += 256) {
    float x = logits[(size_t)b * V_ + v];
    if (x > mx) { s = s * __expf(mx - x) + 1.f; mx = x; }
    else        { s += __expf(x - mx); }
  }
  sx[tid] = mx; sm[tid] = s;
  __syncthreads();
  for (int off = 128; off > 0; off >>= 1) {
    if (tid < off) {
      float m1 = sx[tid], s1 = sm[tid], m2 = sx[tid + off], s2 = sm[tid + off];
      float M = fmaxf(m1, m2);
      sm[tid] = s1 * __expf(m1 - M) + s2 * __expf(m2 - M);
      sx[tid] = M;
    }
    __syncthreads();
  }
  if (tid == 0) { red[b * 2] = sx[0]; red[b * 2 + 1] = sm[0]; }
}

// ---- normalize: write x_t to d_out and bf16 probs for the e_next GEMM ----
__global__ void softmax_norm(const float* __restrict__ logits, const float* __restrict__ red,
                             float* __restrict__ out_t, __bf16* __restrict__ pb) {
  int v = blockIdx.x * blockDim.x + threadIdx.x;   // 125*256 = 32000
#pragma unroll
  for (int b = 0; b < B_; ++b) {
    float M = red[b * 2], invZ = 1.f / red[b * 2 + 1];
    float p = __expf(logits[(size_t)b * V_ + v] - M) * invZ;
    out_t[(size_t)b * (T_ * V_) + v] = p;          // out layout [B,T,V], t folded in
    pb[(size_t)b * V_ + v] = (__bf16)p;
  }
}

// ---- e_acc = p @ emb_in, K=32000 split into 50 deterministic partials ----
__global__ void enext_wmma(const __bf16* __restrict__ pb, const __bf16* __restrict__ embinT,
                           float* __restrict__ partial) {
  int lane = threadIdx.x & 31, wave = threadIdx.x >> 5;
  int ks = blockIdx.x >> 2;                 // K slice 0..49
  int n0 = ((blockIdx.x & 3) * 8 + wave) * 16;  // 32 tiles over E=512
  int mm = lane & 15, hg = lane >> 4;
  int n = n0 + mm;
  v8f c = {};
  int kbase = ks * (KC_PER_SLICE * 32);
  for (int kc = 0; kc < KC_PER_SLICE; ++kc) {
    int k0 = kbase + kc * 32;
    v16bf a = load_a16(pb, V_, mm, hg, k0);
    v16bf b = load_b16(embinT, V_, n, hg, k0);
    c = wmma_bf16(a, b, c);
  }
  if (hg == 0) {                             // rows 0..7 only (B=8)
#pragma unroll
    for (int vg = 0; vg < 8; ++vg)
      partial[ks * (B_ * E_) + vg * E_ + n] = c[vg];
  }
}

// ---- e_t = (sum_ks partial) * img_feats, cast to bf16 --------------------
__global__ void make_et(const float* __restrict__ partial, const float* __restrict__ img,
                        __bf16* __restrict__ etb) {
  int i = blockIdx.x * blockDim.x + threadIdx.x;   // 8*512
  float s = 0.f;
  for (int ks = 0; ks < KSLICES; ++ks) s += partial[ks * (B_ * E_) + i];
  etb[i] = (__bf16)(s * img[i]);                    // img is [16][512], rows 0-7 used
}

// ---- host orchestration --------------------------------------------------
extern "C" void kernel_launch(void* const* d_in, const int* in_sizes, int n_in,
                              void* d_out, int out_size, void* d_ws, size_t ws_size,
                              hipStream_t stream) {
  const float* x       = (const float*)d_in[0];
  const float* enc_w   = (const float*)d_in[1];
  const float* enc_b   = (const float*)d_in[2];
  const float* out_w   = (const float*)d_in[3];
  const float* out_b   = (const float*)d_in[4];
  const float* emb_out = (const float*)d_in[5];
  const float* emb_in  = (const float*)d_in[6];
  const float* ph0     = (const float*)d_in[7];
  const float* w_ih    = (const float*)d_in[8];
  const float* b_ih    = (const float*)d_in[9];
  const float* w_hh    = (const float*)d_in[10];
  const float* b_hh    = (const float*)d_in[11];
  const float* gumbel  = (const float*)d_in[12];
  float* out = (float*)d_out;

  char* ws = (char*)d_ws;
  size_t o = 0;
  auto take = [&](size_t bytes) { char* p = ws + o; o += (bytes + 255) & ~(size_t)255; return p; };
  __bf16* embout_bf = (__bf16*)take((size_t)V_ * H_ * 2);
  __bf16* embinT_bf = (__bf16*)take((size_t)E_ * V_ * 2);
  __bf16* wih_bf    = (__bf16*)take((size_t)G3 * E_ * 2);
  __bf16* whh_bf    = (__bf16*)take((size_t)G3 * H_ * 2);
  float*  h0        = (float*)take(16 * D_ * 4);
  float*  h1        = (float*)take(16 * D_ * 4);
  float*  img       = (float*)take(16 * E_ * 4);
  float*  ht        = (float*)take(16 * H_ * 4);
  __bf16* htb       = (__bf16*)take(16 * H_ * 2);
  __bf16* etb       = (__bf16*)take(16 * E_ * 2);
  float*  gi        = (float*)take(B_ * G3 * 4);
  float*  gh        = (float*)take(B_ * G3 * 4);
  float*  logits    = (float*)take((size_t)B_ * V_ * 4);
  float*  red       = (float*)take(B_ * 2 * 4);
  __bf16* pb        = (__bf16*)take((size_t)16 * V_ * 2);
  float*  partial   = (float*)take((size_t)KSLICES * B_ * E_ * 4);

  // one-time (per launch) weight prep: f32 -> bf16, emb_in transposed
  cvt_bf16<<<(V_ * H_ + 255) / 256, 256, 0, stream>>>(emb_out, embout_bf, V_ * H_);
  tcvt_embin<<<(E_ * V_ + 255) / 256, 256, 0, stream>>>(emb_in, embinT_bf);
  cvt_bf16<<<(G3 * E_ + 255) / 256, 256, 0, stream>>>(w_ih, wih_bf, G3 * E_);
  cvt_bf16<<<(G3 * H_ + 255) / 256, 256, 0, stream>>>(w_hh, whh_bf, G3 * H_);

  // encoder (f32 WMMA)
  pad_x<<<(16 * D_) / 256, 256, 0, stream>>>(x, h0);
  dense_f32_wmma<<<D_ / 128, 256, 0, stream>>>(h0, enc_w, enc_b, h1, D_, D_, 1);
  dense_f32_wmma<<<D_ / 128, 256, 0, stream>>>(h1, enc_w + (size_t)D_ * D_, enc_b + D_, h0, D_, D_, 1);
  dense_f32_wmma<<<E_ / 128, 256, 0, stream>>>(h0, out_w, out_b, img, E_, D_, 0);

  init_state<<<(16 * H_) / 256, 256, 0, stream>>>(ph0, ht, htb, etb);

  for (int t = 0; t < T_; ++t) {
    gru_matmul<<<12, 256, 0, stream>>>(etb, htb, wih_bf, whh_bf, b_ih, b_hh, gi, gh);
    gru_pointwise<<<(B_ * H_) / 256, 256, 0, stream>>>(gi, gh, ht, htb);
    logits_wmma<<<V_ / 128, 256, 0, stream>>>(htb, embout_bf,
                                              gumbel + (size_t)t * B_ * V_, logits);
    softmax_reduce<<<B_, 256, 0, stream>>>(logits, red);
    softmax_norm<<<V_ / 256, 256, 0, stream>>>(logits, red, out + (size_t)t * V_, pb);
    enext_wmma<<<KSLICES * 4, 256, 0, stream>>>(pb, embinT_bf, partial);
    make_et<<<(B_ * E_) / 256, 256, 0, stream>>>(partial, img, etb);
  }
}
